// GluTracker_61186104099741
// MI455X (gfx1250) — matrix operand order repaired and checked by reading, not compiled, not run on hardware
//
#include <hip/hip_runtime.h>

typedef __attribute__((ext_vector_type(16))) _Float16 v16h;
typedef __attribute__((ext_vector_type(8)))  _Float16 v8h;
typedef __attribute__((ext_vector_type(8)))  float    v8f;

union F16x16 { v16h v; v8h h[2]; };

#define H_IN 384
#define W_IN 512
#define H0   96
#define W0   128
#define H1   32
#define W1D  42
#define OC   128
#define KKC  147     // 3*7*7
#define KPAD 160
#define NPTS 8192    // queries per batch
#define NTOT 16384   // B*N

__device__ __forceinline__ int clampi(int v, int lo, int hi) {
    return v < lo ? lo : (v > hi ? hi : v);
}

// ---- weight packing: f32 -> f16, K-contiguous per row (frag-load friendly) ----
// WkA[oc][160]   : conv A-operand rows (zero pad k>=147)
// W1T[n][96]     : MLP1 B-operand rows (zero pad k>=81)
// WpvT[n(16)][256]: MLP2 B-operand rows, cols = [wp(9) | wv | zeros]
__global__ __launch_bounds__(256)
void pack_kernel(const float* __restrict__ conv_w, const float* __restrict__ w1,
                 const float* __restrict__ wp, const float* __restrict__ wv,
                 _Float16* __restrict__ WkA, _Float16* __restrict__ W1T,
                 _Float16* __restrict__ WpvT) {
    int i = blockIdx.x * 256 + threadIdx.x;
    if (i < OC * KPAD) {
        int oc = i / KPAD, k = i % KPAD;
        float v = (k < KKC) ? conv_w[oc * KKC + k] : 0.f;
        WkA[i] = (_Float16)v;
    } else if (i < OC * KPAD + 256 * 96) {
        int j = i - OC * KPAD;
        int n = j / 96, k = j % 96;
        float v = (k < 81) ? w1[k * 256 + n] : 0.f;
        W1T[j] = (_Float16)v;
    } else if (i < OC * KPAD + 256 * 96 + 16 * 256) {
        int j = i - (OC * KPAD + 256 * 96);
        int n = j / 256, k = j % 256;
        float v = 0.f;
        if (n < 9) v = wp[k * 9 + n];
        else if (n == 9) v = wv[k];
        WpvT[j] = (_Float16)v;
    }
}

__global__ __launch_bounds__(256)
void init_kernel(const float* __restrict__ q, float* __restrict__ cur,
                 float* __restrict__ vis) {
    int i = blockIdx.x * 256 + threadIdx.x;
    if (i < NTOT * 2) cur[i] = q[i];
    if (i < NTOT) vis[i] = 0.f;
}

// ------------- conv 7x7 stride4 + bias + ReLU as implicit WMMA GEMM ---------
// grid (8 px-tiles, 96 rows, 4 images), block 256 = 8 waves; wave w -> oc tile w
__global__ __launch_bounds__(256)
void conv_kernel(const float* __restrict__ frames, const float* __restrict__ conv_b,
                 const _Float16* __restrict__ WkA, float* __restrict__ fm) {
    __shared__ _Float16 Bt[16][KPAD];          // patches: 16 pixels x K (row/lane)
    int x0 = blockIdx.x * 16;
    int oy = blockIdx.y;
    int bs = blockIdx.z;
    int tid = threadIdx.x;

    for (int idx = tid; idx < 16 * KPAD; idx += 256) {
        int px = idx / KPAD, k = idx % KPAD;
        float v = 0.f;
        if (k < KKC) {
            int c = k / 49, r = k % 49, ky = r / 7, kx = r % 7;
            int iy = oy * 4 - 3 + ky;
            int ix = (x0 + px) * 4 - 3 + kx;
            if (iy >= 0 && iy < H_IN && ix >= 0 && ix < W_IN) {
                float f = frames[((size_t)(bs * 3 + c) * H_IN + iy) * W_IN + ix];
                v = 2.f * (f * (1.f / 255.f)) - 1.f;
            }
        }
        Bt[px][k] = (_Float16)v;
    }
    __syncthreads();

    int wave = tid >> 5, lane = tid & 31;
    int half = lane >> 4, m = lane & 15;
    int ocb = wave * 16;
    const _Float16* arow = WkA + (size_t)(ocb + m) * KPAD + 16 * half;
    const _Float16* brow = &Bt[m][16 * half];
    v8f acc = {};
    for (int s = 0; s < 5; ++s) {
        F16x16 au, bu;
        au.h[0] = *(const v8h*)(arow + s * 32);        // K = k0+16h .. +7
        au.h[1] = *(const v8h*)(arow + s * 32 + 8);    // K = k0+16h+8 .. +15
        bu.v    = *(const v16h*)(brow + s * 32);       // K = k0+16h .. +15
        acc = __builtin_amdgcn_wmma_f32_16x16x32_f16(false, au.v, false, bu.v,
                                                     (short)0, acc, false, false);
    }
#pragma unroll
    for (int r = 0; r < 8; ++r) {
        int oc = ocb + r + 8 * half;
        float v = acc[r] + conv_b[oc];
        v = v > 0.f ? v : 0.f;
        fm[((size_t)(bs * OC + oc) * H0 + oy) * W0 + (x0 + m)] = v;
    }
}

// ---------------------------- 3x3 avg pool s3 -------------------------------
__global__ __launch_bounds__(256)
void pool_kernel(const float* __restrict__ fm, float* __restrict__ pyr) {
    int i = blockIdx.x * 256 + threadIdx.x;
    if (i >= 4 * OC * H1 * W1D) return;
    int x = i % W1D; int t = i / W1D;
    int y = t % H1;  t /= H1;
    int c = t % OC;  int bs = t / OC;
    const float* src = fm + (size_t)(bs * OC + c) * H0 * W0;
    float s = 0.f;
#pragma unroll
    for (int dy = 0; dy < 3; ++dy)
#pragma unroll
        for (int dx = 0; dx < 3; ++dx)
            s += src[(y * 3 + dy) * W0 + (x * 3 + dx)];
    pyr[i] = s * (1.f / 9.f);
}

// ------------- fused per-level tracker: sample+norm+corr+MLP+update ---------
// block 256 = 8 waves = 8 points; grid 2048
__global__ __launch_bounds__(256)
void track_kernel(const float* __restrict__ fmap, int hh, int ww,
                  const _Float16* __restrict__ W1T, const float* __restrict__ b1,
                  const _Float16* __restrict__ WpvT, const float* __restrict__ bp,
                  const float* __restrict__ bv,
                  float* __restrict__ cur, float* __restrict__ vis) {
    __shared__ _Float16 fnS[8][9][128];   // corr A rows: [h][c]
    __shared__ _Float16 mnS[8][9][128];   // corr B rows: [col][c] (lane = column)
    __shared__ _Float16 corrS[8][96];     // MLP1 A rows (K padded to 96)
    __shared__ _Float16 hS[8][256];       // MLP2 A rows
    __shared__ float    logitsS[8][16];

    int tid = threadIdx.x;
    int wave = tid >> 5, lane = tid & 31;
    int half = lane >> 4, m = lane & 15;
    int pt = wave;
    size_t gp = (size_t)blockIdx.x * 8 + pt;
    int bidx = (int)(gp >> 13);           // / NPTS
    float px = cur[gp * 2 + 0];
    float py = cur[gp * 2 + 1];
    const v8h zero8 = {};

    if (lane < 15) corrS[pt][81 + lane] = (_Float16)0.f;   // K pad

    // bilinear sampling of 3x3 window, both frames, L2-normalized over C
    for (int sfr = 0; sfr < 2; ++sfr) {
        const float* base = fmap + (size_t)(bidx * 2 + sfr) * OC * hh * ww;
        for (int wpos = 0; wpos < 9; ++wpos) {
            int wi = wpos / 3, wj = wpos % 3;
            // align_corners grid algebra collapses to pixel coords +/- 1
            float ix = px * (float)(ww - 1) + (float)(wj - 1);
            float iy = py * (float)(hh - 1) + (float)(wi - 1);
            float x0f = floorf(ix), y0f = floorf(iy);
            float wx1 = ix - x0f, wx0 = 1.f - wx1;
            float wy1 = iy - y0f, wy0 = 1.f - wy1;
            int X0 = (int)x0f, Y0 = (int)y0f, X1 = X0 + 1, Y1 = Y0 + 1;
            bool vX0 = X0 >= 0 && X0 < ww, vX1 = X1 >= 0 && X1 < ww;
            bool vY0 = Y0 >= 0 && Y0 < hh, vY1 = Y1 >= 0 && Y1 < hh;
            int cX0 = clampi(X0, 0, ww - 1), cX1 = clampi(X1, 0, ww - 1);
            int cY0 = clampi(Y0, 0, hh - 1), cY1 = clampi(Y1, 0, hh - 1);
            float w00 = wx0 * wy0 * (float)(vX0 && vY0);
            float w10 = wx1 * wy0 * (float)(vX1 && vY0);
            float w01 = wx0 * wy1 * (float)(vX0 && vY1);
            float w11 = wx1 * wy1 * (float)(vX1 && vY1);

            float vals[4]; float ss = 0.f;
#pragma unroll
            for (int cb = 0; cb < 4; ++cb) {
                int c = cb * 32 + lane;
                const float* fc = base + (size_t)c * hh * ww;
                float v = w00 * fc[cY0 * ww + cX0] + w10 * fc[cY0 * ww + cX1]
                        + w01 * fc[cY1 * ww + cX0] + w11 * fc[cY1 * ww + cX1];
                vals[cb] = v;
                ss += v * v;
            }
#pragma unroll
            for (int off = 16; off >= 1; off >>= 1) ss += __shfl_xor(ss, off, 32);
            float scale = 1.f / fmaxf(sqrtf(ss), 1e-12f);
#pragma unroll
            for (int cb = 0; cb < 4; ++cb) {
                int c = cb * 32 + lane;
                _Float16 hv = (_Float16)(vals[cb] * scale);
                if (sfr == 0) fnS[pt][wpos][c] = hv;
                else          mnS[pt][wpos][c] = hv;
            }
        }
    }
    asm volatile("s_wait_dscnt 0" ::: "memory");  // LDS stores visible to wave
    __syncthreads();

    // corr: 9x9 = fn^T(9x128) @ mn(128x9), K=128 in 4 WMMA steps (per wave)
    {
        const _Float16* ar = &fnS[pt][m < 9 ? m : 0][16 * half];
        const _Float16* br = &mnS[pt][m < 9 ? m : 0][16 * half];
        v8f acc = {};
        for (int s = 0; s < 4; ++s) {
            F16x16 au, bu;
            if (m < 9) {
                au.h[0] = *(const v8h*)(ar + s * 32);
                au.h[1] = *(const v8h*)(ar + s * 32 + 8);
                bu.v    = *(const v16h*)(br + s * 32);
            } else {
                au.h[0] = zero8; au.h[1] = zero8;
                bu.h[0] = zero8; bu.h[1] = zero8;
            }
            acc = __builtin_amdgcn_wmma_f32_16x16x32_f16(false, au.v, false, bu.v,
                                                         (short)0, acc, false, false);
        }
#pragma unroll
        for (int r = 0; r < 8; ++r) {
            int hrow = r + 8 * half;
            if (hrow < 9 && m < 9) corrS[pt][hrow * 9 + m] = (_Float16)acc[r];
        }
    }
    __syncthreads();

    // MLP layer 1: (8 pts x 96) @ (96 x 256), ReLU(+b1); wave handles 2 N-tiles
    for (int t = 0; t < 2; ++t) {
        int n0 = (wave * 2 + t) * 16;
        const _Float16* ar = &corrS[m < 8 ? m : 0][16 * half];
        const _Float16* br = WpvT; // placeholder, reassigned below
        br = W1T + (size_t)(n0 + m) * 96 + 16 * half;
        v8f acc = {};
        for (int s = 0; s < 3; ++s) {
            F16x16 au, bu;
            if (m < 8) {
                au.h[0] = *(const v8h*)(ar + s * 32);
                au.h[1] = *(const v8h*)(ar + s * 32 + 8);
            } else { au.h[0] = zero8; au.h[1] = zero8; }
            bu.v = *(const v16h*)(br + s * 32);
            acc = __builtin_amdgcn_wmma_f32_16x16x32_f16(false, au.v, false, bu.v,
                                                         (short)0, acc, false, false);
        }
        if (half == 0) {
#pragma unroll
            for (int r = 0; r < 8; ++r) {
                float v = acc[r] + b1[n0 + m];
                hS[r][n0 + m] = (_Float16)fmaxf(v, 0.f);
            }
        }
    }
    __syncthreads();

    // MLP layer 2: (8 pts x 256) @ (256 x [9 pred | 1 vis | pad])
    if (wave == 0) {
        const _Float16* ar = &hS[m < 8 ? m : 0][16 * half];
        const _Float16* br = WpvT + (size_t)m * 256 + 16 * half;
        v8f acc = {};
        for (int s = 0; s < 8; ++s) {
            F16x16 au, bu;
            if (m < 8) {
                au.h[0] = *(const v8h*)(ar + s * 32);
                au.h[1] = *(const v8h*)(ar + s * 32 + 8);
            } else { au.h[0] = zero8; au.h[1] = zero8; }
            bu.v = *(const v16h*)(br + s * 32);
            acc = __builtin_amdgcn_wmma_f32_16x16x32_f16(false, au.v, false, bu.v,
                                                         (short)0, acc, false, false);
        }
        if (half == 0) {
#pragma unroll
            for (int r = 0; r < 8; ++r) logitsS[r][m] = acc[r];
        }
    }
    __syncthreads();

    // epilogue: softmax soft-argmax + sigmoid vis, position update
    if (tid < 8) {
        int p = tid;
        size_t g = (size_t)blockIdx.x * 8 + p;
        float l[9], mx = -1e30f;
#pragma unroll
        for (int i = 0; i < 9; ++i) { l[i] = logitsS[p][i] + bp[i]; mx = fmaxf(mx, l[i]); }
        float sum = 0.f;
#pragma unroll
        for (int i = 0; i < 9; ++i) { l[i] = expf(l[i] - mx); sum += l[i]; }
        float inv = 1.f / sum;
        float d_x = 0.f, d_y = 0.f;
#pragma unroll
        for (int i = 0; i < 3; ++i) {     // pred.sum(-1) @ dsx   (dsx uses ww)
            float rs = (l[i * 3] + l[i * 3 + 1] + l[i * 3 + 2]) * inv;
            d_x += rs * (float)(i - 1) / (float)(ww - 1);
        }
#pragma unroll
        for (int j = 0; j < 3; ++j) {     // pred.sum(-2) @ dsy   (dsy uses hh)
            float cs = (l[j] + l[3 + j] + l[6 + j]) * inv;
            d_y += cs * (float)(j - 1) / (float)(hh - 1);
        }
        float vl = logitsS[p][9] + bv[0];
        cur[g * 2 + 0] += d_x;
        cur[g * 2 + 1] += d_y;
        vis[g] += 1.f / (1.f + expf(-vl));
    }
}

__global__ __launch_bounds__(256)
void fin_kernel(const float* __restrict__ cur, const float* __restrict__ vis,
                float* __restrict__ out) {
    int i = blockIdx.x * 256 + threadIdx.x;
    if (i < NTOT * 2) out[i] = cur[i];
    if (i < NTOT) out[NTOT * 2 + i] = vis[i] * 0.5f;   // / N_LVL
}

extern "C" void kernel_launch(void* const* d_in, const int* in_sizes, int n_in,
                              void* d_out, int out_size, void* d_ws, size_t ws_size,
                              hipStream_t stream) {
    const float* queries = (const float*)d_in[0];
    const float* frames  = (const float*)d_in[1];
    const float* conv_w  = (const float*)d_in[2];
    const float* conv_b  = (const float*)d_in[3];
    const float* w1      = (const float*)d_in[4];
    const float* b1      = (const float*)d_in[5];
    const float* wp      = (const float*)d_in[6];
    const float* bp      = (const float*)d_in[7];
    const float* wv      = (const float*)d_in[8];
    const float* bv      = (const float*)d_in[9];

    char* ws = (char*)d_ws;
    _Float16* WkA  = (_Float16*)(ws + 0);         // 128*160*2   = 40960
    _Float16* W1T  = (_Float16*)(ws + 40960);     // 256*96*2    = 49152
    _Float16* WpvT = (_Float16*)(ws + 90112);     // 16*256*2    = 8192
    float* cur     = (float*)(ws + 98304);        // 32768*4     = 131072
    float* vis     = (float*)(ws + 229376);       // 16384*4     = 65536
    float* fm      = (float*)(ws + 294912);       // 4*128*96*128*4 = 25165824
    float* pyr     = (float*)(ws + 25460736);     // 4*128*32*42*4  = 2752512

    pack_kernel<<<192, 256, 0, stream>>>(conv_w, w1, wp, wv, WkA, W1T, WpvT);
    init_kernel<<<128, 256, 0, stream>>>(queries, cur, vis);
    conv_kernel<<<dim3(8, 96, 4), 256, 0, stream>>>(frames, conv_b, WkA, fm);
    pool_kernel<<<2688, 256, 0, stream>>>(fm, pyr);
    // coarse level (pyr[1]) then fine level (pyr[0]); positions chain through ws
    track_kernel<<<2048, 256, 0, stream>>>(pyr, H1, W1D, W1T, b1, WpvT, bp, bv, cur, vis);
    track_kernel<<<2048, 256, 0, stream>>>(fm,  H0, W0,  W1T, b1, WpvT, bp, bv, cur, vis);
    fin_kernel<<<192, 256, 0, stream>>>(cur, vis, (float*)d_out);
}
